// MyLAB_54434415510126
// MI455X (gfx1250) — compile-verified, matrix operand order checked by reading, not run on hardware
//
#include <hip/hip_runtime.h>
#include <hip/hip_bf16.h>

typedef __attribute__((ext_vector_type(16))) _Float16 v16h;
typedef __attribute__((ext_vector_type(8)))  _Float16 v8h;
typedef __attribute__((ext_vector_type(8)))  float    v8f;
typedef __attribute__((ext_vector_type(4))) unsigned int v4u;
typedef __attribute__((ext_vector_type(8))) int v8i;
typedef __attribute__((ext_vector_type(4))) int v4i;

#define ED   64
#define FD   256
#define PP   64          // PH*PW
#define LLN  4096        // Hn*Wn
#define BBN  4
#define CIN  32
#define HH   512
#define WWN  512
#define HWS  (HH*WWN)
#define NBLK 2

#if defined(__gfx1250__) && __has_builtin(__builtin_amdgcn_tensor_load_to_lds) && \
    __has_builtin(__builtin_amdgcn_s_wait_tensorcnt)
#define HAVE_TDM 1
#else
#define HAVE_TDM 0
#endif

__device__ __forceinline__ v8f wmma16(v16h a, v16h b, v8f c) {
  return __builtin_amdgcn_wmma_f32_16x16x32_f16(false, a, false, b, (short)0, c, false, false);
}

// 16x32 f16 operand fragment from LDS, row-major [row][k], ld in halfs.
// wave32 mapping: lane L -> row = row0+(L&15); K-halves split by L>>4.
__device__ __forceinline__ v16h frag_lds(const _Float16* base, int row0, int ld, int k0) {
  int lane = threadIdx.x & 31;
  const _Float16* p = base + (row0 + (lane & 15)) * ld + k0 + (lane >> 4) * 8;
  v8h a = *(const v8h*)(p);
  v8h b = *(const v8h*)(p + 16);
  v16h o;
#pragma unroll
  for (int i = 0; i < 8; ++i) { o[i] = a[i]; o[i + 8] = b[i]; }
  return o;
}

// Fragment built from f32 global weights [row][k] (converted to f16).
__device__ __forceinline__ v16h frag_gf32(const float* base, int row0, int ld, int k0) {
  int lane = threadIdx.x & 31;
  const float* p = base + (size_t)(row0 + (lane & 15)) * ld + k0 + (lane >> 4) * 8;
  v16h o;
#pragma unroll
  for (int i = 0; i < 8; ++i) { o[i] = (_Float16)p[i]; o[i + 8] = (_Float16)p[16 + i]; }
  return o;
}

#if HAVE_TDM
__device__ __forceinline__ uint32_t lds_off32(const void* p) {
  return (uint32_t)(uintptr_t)(__attribute__((address_space(3))) const void*)p;
}
// TDM: load a [rows x 64-half] tile (contiguous 128B rows) from global into LDS,
// inserting 16B padding every 128B so the LDS leading dim is 72 halfs.
// D# per CDNA5 ISA. Issue from ONE wave only.
__device__ __forceinline__ void tdm_load_tile(uint32_t ldsoff, const _Float16* gsrc, int rows) {
  uint64_t ga = (uint64_t)(uintptr_t)gsrc;
  v4u g0;
  g0[0] = 1u;                                           // count=1, user mode
  g0[1] = ldsoff;                                       // lds_addr
  g0[2] = (uint32_t)ga;                                 // global_addr[31:0]
  g0[3] = (uint32_t)((ga >> 32) & 0x1ffffffu) | (2u << 30); // addr[56:32] | type=2
  v8i g1;
  g1[0] = (1 << 16) | (1 << 20) | (4 << 22) | (3 << 25); // data_size=2B, pad_en, 32dw interval, 4dw pad
  g1[1] = (64 << 16);                                    // tensor_dim0 = 64 (lo16)
  g1[2] = (rows << 16);                                  // tensor_dim0 hi=0 | tensor_dim1 lo16
  g1[3] = (64 << 16);                                    // tensor_dim1 hi=0 | tile_dim0 = 64
  g1[4] = rows;                                          // tile_dim1 = rows, tile_dim2 = 0
  g1[5] = 64;                                            // tensor_dim0_stride = 64 (lo32)
  g1[6] = 0;
  g1[7] = 0;
  v4i gz4 = {0, 0, 0, 0};
  v8i gz8 = {0, 0, 0, 0, 0, 0, 0, 0};
  // 6-arg toolchain signature: (v4u, v8i, v4i, v4i, v8i, i32 cpol)
  __builtin_amdgcn_tensor_load_to_lds(g0, g1, gz4, gz4, gz8, 0);
}
#endif

// ---------------- Stage 0: 1x1 conv 32->64 + relu + dw affine, patchify ----------------
__global__ void __launch_bounds__(256)
k_proj(const float* __restrict__ x, const float* __restrict__ w1,
       const float* __restrict__ b1, const float* __restrict__ dww,
       const float* __restrict__ dwb, _Float16* __restrict__ ps)
{
  __shared__ alignas(16) _Float16 xb[256 * 40];   // [n][c] f16, padded
  __shared__ float sb1[64], sdw[64], sdb[64];

  int tid = threadIdx.x;
  int pix0 = blockIdx.x * 256;           // 256 consecutive pixels, same image row
  int b    = pix0 / HWS;
  int rem  = pix0 % HWS;
  int h    = rem / WWN;
  int w0   = rem % WWN;

  if (tid < 64) { sb1[tid] = b1[tid]; sdw[tid] = dww[tid]; sdb[tid] = dwb[tid]; }
  const float* xbase = x + (size_t)b * CIN * HWS + (size_t)h * WWN + w0;
#pragma unroll 4
  for (int c = 0; c < CIN; ++c)
    xb[tid * 40 + c] = (_Float16)xbase[(size_t)c * HWS + tid];
  __syncthreads();

  int wave = tid >> 5, lane = tid & 31;
#pragma unroll 2
  for (int t = wave; t < 4 * 16; t += 8) {       // 4 M-tiles x 16 N-tiles
    int mt = t & 3, nt = t >> 2;
    v16h a  = frag_gf32(w1, mt * 16, 32, 0);     // proj_w1 (64,32), loop-invariant per mt
    v16h bf = frag_lds(xb, nt * 16, 40, 0);
    v8f acc = {};
    acc = wmma16(a, bf, acc);

    int n  = nt * 16 + (lane & 15);
    int mb = mt * 16 + (lane >> 4) * 8;
    int wc = w0 + n;
    int p  = (h & 7) * 8 + (wc & 7);
    int l  = (h >> 3) * 64 + (wc >> 3);
    v8h o;
#pragma unroll
    for (int r = 0; r < 8; ++r) {
      int m = mb + r;
      float v = acc[r] + sb1[m];
      v = fmaxf(v, 0.f);
      o[r] = (_Float16)(v * sdw[m] + sdb[m]);
    }
    *(v8h*)(ps + (((size_t)(b * PP + p)) * LLN + l) * ED + mb) = o;
  }
}

// ---------------- Per-block pass 1: cv[b,oc,p] = sum_l x2*x1 ----------------
__global__ void __launch_bounds__(256)
k_cv(const _Float16* __restrict__ ps, const float* __restrict__ ipw,
     const float* __restrict__ ipb, float* __restrict__ cv)
{
  __shared__ alignas(16) _Float16 pb[128 * 72];   // ps tile [n][64] (+pad)
  __shared__ alignas(16) _Float16 wA[80 * 72];    // ip_w rows 0..79, f16
  __shared__ float tb[80 * 132];                  // t tile f32 [m][n]
  __shared__ float sbias[80];

  int g = blockIdx.x;                    // b*64 + p
  size_t gbase = (size_t)g * LLN * ED;
  int tid = threadIdx.x, wave = tid >> 5, lane = tid & 31;

  for (int i = tid; i < 80 * 64; i += 256) {
    int m = i >> 6, k = i & 63;
    wA[m * 72 + k] = (_Float16)ipw[i];
  }
  if (tid < 80) sbias[tid] = ipb[tid];
  float acc = 0.f;
  int oc = tid >> 2, q = tid & 3;

  for (int l0 = 0; l0 < LLN; l0 += 128) {
    __syncthreads();
#if HAVE_TDM
    if (wave == 0) {
      tdm_load_tile(lds_off32(pb), ps + gbase + (size_t)l0 * ED, 128);
      __builtin_amdgcn_s_wait_tensorcnt(0);
    }
#else
    for (int i = tid; i < 128 * 8; i += 256) {
      int n = i >> 3, c8 = i & 7;
      *(v8h*)(pb + n * 72 + c8 * 8) =
          *(const v8h*)(ps + gbase + (size_t)(l0 + n) * ED + c8 * 8);
    }
#endif
    __syncthreads();
    int nt = wave;                        // 8 waves -> 8 N-tiles of 16
#pragma unroll
    for (int mt = 0; mt < 5; ++mt) {      // rows 0..79 of ip_w (0..64 used)
      v8f a2 = {};
      a2 = wmma16(frag_lds(wA, mt * 16, 72, 0),  frag_lds(pb, nt * 16, 72, 0),  a2);
      a2 = wmma16(frag_lds(wA, mt * 16, 72, 32), frag_lds(pb, nt * 16, 72, 32), a2);
      int n  = nt * 16 + (lane & 15);
      int mb = mt * 16 + (lane >> 4) * 8;
#pragma unroll
      for (int r = 0; r < 8; ++r)
        tb[(mb + r) * 132 + n] = a2[r] + sbias[mb + r];
    }
    __syncthreads();
#pragma unroll 4
    for (int n = q * 32; n < q * 32 + 32; ++n)
      acc += tb[(1 + oc) * 132 + n] * tb[n];       // x2[oc]*x1
  }
  __syncthreads();
  tb[tid] = acc;                                   // tid == oc*4+q
  __syncthreads();
  if (tid < 64)
    cv[g * 64 + tid] = tb[tid * 4] + tb[tid * 4 + 1] + tb[tid * 4 + 2] + tb[tid * 4 + 3];
}

// ---------------- Per-block pass 2: x3 -> modulate -> ff1 -> ff2 ----------------
__global__ void __launch_bounds__(256)
k_block(const float* __restrict__ ipw, const float* __restrict__ ipb,
        const float* __restrict__ cv,
        const float* __restrict__ opw, const float* __restrict__ opb,
        const float* __restrict__ f1w, const float* __restrict__ f1b,
        const float* __restrict__ f2w, const float* __restrict__ f2b,
        _Float16* __restrict__ ps, float* __restrict__ out, int last)
{
  __shared__ alignas(16) _Float16 pb[64 * 72];    // ps tile  [n][64]
  __shared__ alignas(16) _Float16 ob[64 * 72];    // o tile   [n][64]
  __shared__ alignas(16) _Float16 hb[64 * 264];   // h1 tile  [n][256]
  __shared__ alignas(16) _Float16 w3s[64 * 72];   // ip_w rows 65..128, f16
  __shared__ alignas(16) _Float16 wf1s[256 * 72]; // ff_w1 f16
  __shared__ alignas(16) _Float16 wf2s[64 * 264]; // ff_w2 f16
  __shared__ float sb3[64], sf1b[256], sf2b[64], sA[64], sB[64];

  int tid = threadIdx.x, wave = tid >> 5, lane = tid & 31;
  const float* w3 = ipw + 65 * 64;                // x3 rows of ip_w

  for (int i = tid; i < 64 * 64; i += 256) {
    int m = i >> 6, k = i & 63;
    w3s[m * 72 + k] = (_Float16)w3[i];
  }
  for (int i = tid; i < 256 * 64; i += 256) {
    int m = i >> 6, k = i & 63;
    wf1s[m * 72 + k] = (_Float16)f1w[i];
  }
  for (int i = tid; i < 64 * 256; i += 256) {
    int m = i >> 8, k = i & 255;
    wf2s[m * 264 + k] = (_Float16)f2w[i];
  }
  if (tid < 64) { sb3[tid] = ipb[65 + tid]; sf2b[tid] = f2b[tid]; sB[tid] = opb[tid]; }
  sf1b[tid] = f1b[tid];

  const int NCH = BBN * PP * LLN / 64;            // 64-position chunks
  for (int ch = blockIdx.x; ch < NCH; ch += gridDim.x) {
    __syncthreads();
    size_t pos0 = (size_t)ch * 64;
    int g = (int)(pos0 / LLN);                    // (b,p) group
#if HAVE_TDM
    if (wave == 0) {
      tdm_load_tile(lds_off32(pb), ps + pos0 * ED, 64);
      __builtin_amdgcn_s_wait_tensorcnt(0);
    }
#else
    for (int i = tid; i < 64 * 8; i += 256) {
      int n = i >> 3, c8 = i & 7;
      *(v8h*)(pb + n * 72 + c8 * 8) = *(const v8h*)(ps + (pos0 + n) * ED + c8 * 8);
    }
#endif
    if (tid < 64) sA[tid] = cv[g * 64 + tid] * opw[tid];
    // prefetch next chunk's ps tile (global_prefetch_b8)
    if (ch + (int)gridDim.x < NCH && tid < 64)
      __builtin_prefetch(ps + ((size_t)(ch + gridDim.x) * 64) * ED + tid * 64, 0, 1);
    __syncthreads();

    // GEMM1: x3 (64x64, K=64) -> ob = relu(x3)*cv*opw + opb
#pragma unroll 2
    for (int t = wave; t < 16; t += 8) {
      int mt = t & 3, nt = t >> 2;
      v8f a2 = {};
      a2 = wmma16(frag_lds(w3s, mt * 16, 72, 0),  frag_lds(pb, nt * 16, 72, 0),  a2);
      a2 = wmma16(frag_lds(w3s, mt * 16, 72, 32), frag_lds(pb, nt * 16, 72, 32), a2);
      int n = nt * 16 + (lane & 15), mb = mt * 16 + (lane >> 4) * 8;
      v8h o;
#pragma unroll
      for (int r = 0; r < 8; ++r) {
        float v = fmaxf(a2[r] + sb3[mb + r], 0.f);
        o[r] = (_Float16)(v * sA[mb + r] + sB[mb + r]);
      }
      *(v8h*)(ob + n * 72 + mb) = o;
    }
    __syncthreads();

    // GEMM2: h1 = relu(ff_w1 @ o + b1)   (256x64, K=64)
#pragma unroll 2
    for (int t = wave; t < 64; t += 8) {
      int mt = t & 15, nt = t >> 4;
      v8f a2 = {};
      a2 = wmma16(frag_lds(wf1s, mt * 16, 72, 0),  frag_lds(ob, nt * 16, 72, 0),  a2);
      a2 = wmma16(frag_lds(wf1s, mt * 16, 72, 32), frag_lds(ob, nt * 16, 72, 32), a2);
      int n = nt * 16 + (lane & 15), mb = mt * 16 + (lane >> 4) * 8;
      v8h o;
#pragma unroll
      for (int r = 0; r < 8; ++r)
        o[r] = (_Float16)fmaxf(a2[r] + sf1b[mb + r], 0.f);
      *(v8h*)(hb + n * 264 + mb) = o;
    }
    __syncthreads();

    // GEMM3: ps' = ff_w2 @ h1 + b2   (64x64, K=256)
#pragma unroll 2
    for (int t = wave; t < 16; t += 8) {
      int mt = t & 3, nt = t >> 2;
      v8f a2 = {};
#pragma unroll
      for (int k0 = 0; k0 < 256; k0 += 32)
        a2 = wmma16(frag_lds(wf2s, mt * 16, 264, k0), frag_lds(hb, nt * 16, 264, k0), a2);
      int n = nt * 16 + (lane & 15), mb = mt * 16 + (lane >> 4) * 8;
      size_t pos = pos0 + n;
      if (!last) {
        v8h o;
#pragma unroll
        for (int r = 0; r < 8; ++r) o[r] = (_Float16)(a2[r] + sf2b[mb + r]);
        *(v8h*)(ps + pos * ED + mb) = o;
      } else {
        int b = (int)(pos / (PP * LLN));
        int rm = (int)(pos % (PP * LLN));
        int p = rm / LLN, l = rm % LLN;
        int hh = (l >> 6) * 8 + (p >> 3);
        int ww = (l & 63) * 8 + (p & 7);
        float* dst = out + ((size_t)b * ED + mb) * HWS + (size_t)hh * WWN + ww;
#pragma unroll
        for (int r = 0; r < 8; ++r) dst[(size_t)r * HWS] = a2[r] + sf2b[mb + r];
      }
    }
  }
}

extern "C" void kernel_launch(void* const* d_in, const int* in_sizes, int n_in,
                              void* d_out, int out_size, void* d_ws, size_t ws_size,
                              hipStream_t stream)
{
  (void)in_sizes; (void)n_in; (void)out_size; (void)ws_size;
  const float* x   = (const float*)d_in[0];
  const float* pw1 = (const float*)d_in[1];
  const float* pb1 = (const float*)d_in[2];
  const float* dww = (const float*)d_in[3];
  const float* dwb = (const float*)d_in[4];
  const float* ipw = (const float*)d_in[5];   // (2,129,64)
  const float* ipb = (const float*)d_in[6];   // (2,129)
  const float* opw = (const float*)d_in[7];   // (2,64)
  const float* opb = (const float*)d_in[8];
  const float* f1w = (const float*)d_in[9];   // (2,256,64)
  const float* f1b = (const float*)d_in[10];  // (2,256)
  const float* f2w = (const float*)d_in[11];  // (2,64,256)
  const float* f2b = (const float*)d_in[12];  // (2,64)
  float* out = (float*)d_out;

  _Float16* ps = (_Float16*)d_ws;                       // 4*64*4096*64 f16 = 128 MB
  float* cv = (float*)((char*)d_ws + (size_t)BBN * PP * LLN * ED * sizeof(_Float16));

  k_proj<<<BBN * HWS / 256, 256, 0, stream>>>(x, pw1, pb1, dww, dwb, ps);
  for (int i = 0; i < NBLK; ++i) {
    k_cv<<<BBN * PP, 256, 0, stream>>>(ps, ipw + i * 129 * 64, ipb + i * 129, cv);
    k_block<<<2048, 256, 0, stream>>>(ipw + i * 129 * 64, ipb + i * 129, cv,
                                      opw + i * 64, opb + i * 64,
                                      f1w + i * 256 * 64, f1b + i * 256,
                                      f2w + i * 64 * 256, f2b + i * 64,
                                      ps, out, i == NBLK - 1);
  }
}
